// Decomp_Forecast_10282151706857
// MI455X (gfx1250) — matrix operand order repaired and checked by reading, not compiled, not run on hardware
//
#include <hip/hip_runtime.h>
#include <hip/hip_bf16.h>

typedef _Float16 v16h __attribute__((ext_vector_type(16)));
typedef _Float16 v8h  __attribute__((ext_vector_type(8)));
typedef _Float16 h2   __attribute__((ext_vector_type(2)));
typedef float    v8f  __attribute__((ext_vector_type(8)));

#define BATCH 16
#define SEQ   720
#define CH    321
#define NORD  64
#define NP    (BATCH*CH)   // 5136
#define KPAD  736          // 720 padded to multiple of 32 (K chunk)
#define MPAD  5184         // 81*64

// ---------------- workspace layout (bytes) ----------------
static const size_t OFF_STATS = 0;                        // 5136 * 8 f32
static const size_t OFF_V     = 164352;                   // 720*64 f32 (V[j] = A^j B)
static const size_t OFF_R     = 348672;                   // 720 f32 (row sums of KK)
static const size_t OFF_C     = 351552;                   // 720 f32 (eval @ b_mlp)
static const size_t OFF_KK    = 354432;                   // 720*736 f16, row-major [tau][t]
static const size_t OFF_X     = 1414272;                  // 5184*736 f16, row-major [p][t]
// total ~9.05 MB

// ---------------- kernel 1: per-pair stats ----------------
__global__ void stats_kernel(const float* __restrict__ x, const float* __restrict__ aw,
                             const float* __restrict__ ab, float* __restrict__ stats) {
  int b = blockIdx.x;
  int ch = blockIdx.y * 256 + threadIdx.x;
  if (ch >= CH) return;
  const float* p = x + (size_t)b * SEQ * CH + ch;
  float s = 0.f, sq = 0.f;
  for (int t = 0; t < SEQ; ++t) { float v = p[(size_t)t * CH]; s += v; sq += v * v; }
  float mean = s * (1.f / SEQ);
  float var  = sq * (1.f / SEQ) - mean * mean;
  var = var < 0.f ? 0.f : var;
  float sd   = sqrtf(var + 1e-5f);
  float rstd = 1.f / sd;
  float w = aw[ch], bb = ab[ch];
  float alpha = rstd * w;
  float beta  = bb - mean * alpha;
  float* o = stats + (size_t)(b * CH + ch) * 8;
  o[0] = mean; o[1] = sd; o[2] = alpha; o[3] = beta;
  o[4] = bb;   o[5] = 1.f / (w + 1e-10f); o[6] = 0.f; o[7] = 0.f;
}

// ------- kernel 2: transpose x (b,t,ch) -> Xf16[p][t], zero-pad K to 736 -------
__global__ void xcast_kernel(const float* __restrict__ x, _Float16* __restrict__ X) {
  __shared__ _Float16 sm[32][68];
  int b = blockIdx.x, t0 = blockIdx.y * 32, c0 = blockIdx.z * 64;
  int tid = threadIdx.x;
  for (int e = tid; e < 2048; e += 256) {
    int tl = e >> 6, cl = e & 63;
    int t = t0 + tl, c = c0 + cl;
    float v = (t < SEQ && c < CH) ? x[((size_t)b * SEQ + t) * CH + c] : 0.f;
    sm[tl][cl] = (_Float16)v;
  }
  __syncthreads();
  for (int e = tid; e < 1024; e += 256) {
    int cl = e >> 4, tp = e & 15;
    int c = c0 + cl;
    if (c < CH) {
      int pr = b * CH + c;
      h2 v; v[0] = sm[2 * tp][cl]; v[1] = sm[2 * tp + 1][cl];
      *reinterpret_cast<h2*>(X + (size_t)pr * KPAD + t0 + 2 * tp) = v;
    }
  }
}

// ------- kernel 3: Krylov doubling: V[j] = A^j B, j = 0..719 -------
// A^m kept transposed in LDS (double-buffered, pad 65 -> conflict-free);
// V rows in global (kk_kernel needs them), accessed as same-address broadcasts.
__global__ __launch_bounds__(1024) void krylov_kernel(const float* __restrict__ A,
                                                      const float* __restrict__ Bv,
                                                      float* __restrict__ V) {
  __shared__ float Pt0[64][65];   // Pt[k][r] = (A^m)[r][k]
  __shared__ float Pt1[64][65];
  int tid = threadIdx.x;
  int nt = blockDim.x;
  for (int i = tid; i < 64; i += nt) V[i] = Bv[i];
  for (int o = tid; o < 4096; o += nt) {
    int r = o >> 6, k = o & 63;            // consecutive lanes -> consecutive k
    Pt0[k][r] = A[o];                       // stride-65 rows -> conflict-free
  }
  __threadfence(); __syncthreads();
  bool flip = false;
  for (int m = 1; m < SEQ; m <<= 1) {
    float (*Pr)[65] = flip ? Pt1 : Pt0;
    float (*Pw)[65] = flip ? Pt0 : Pt1;
    int cnt = (m < SEQ - m) ? m : (SEQ - m);
    // V[m+j][r] = sum_k (A^m)[r][k] * V[j][k]
    for (int o = tid; o < cnt * 64; o += nt) {
      int j = o >> 6, r = o & 63;          // lanes share j -> V[j][k] broadcast
      const float* vj = V + (size_t)j * 64;
      float s = 0.f;
      #pragma unroll
      for (int k = 0; k < 64; ++k) s += Pr[k][r] * vj[k];
      V[(size_t)(m + j) * 64 + r] = s;     // lanes consecutive in r -> coalesced
    }
    if (2 * m < SEQ) { // (A^2m)[r][cc] = sum_k (A^m)[r][k]*(A^m)[k][cc]
      for (int o = tid; o < 4096; o += nt) {
        int r = o >> 6, cc = o & 63;       // lanes share r
        float s = 0.f;
        #pragma unroll
        for (int k = 0; k < 64; ++k) s += Pr[k][r] * Pr[cc][k];
        Pw[cc][r] = s;                     // store transposed for next stage
      }
      flip = !flip;
    }
    __threadfence(); __syncthreads();
  }
}

// ------- kernel 4: KK[tau][t] = (eval[tau]·W)·V[719-t]; R[tau]; c[tau] -------
__global__ void kk_kernel(const float* __restrict__ ev, const float* __restrict__ W,
                          const float* __restrict__ bml, const float* __restrict__ V,
                          _Float16* __restrict__ KK, float* __restrict__ R,
                          float* __restrict__ Cc) {
  __shared__ float ew[64];
  __shared__ float red[256];
  int tau = blockIdx.x, tid = threadIdx.x;
  const float* evr = ev + (size_t)tau * 64;
  if (tid < 64) {
    float s = 0.f;
    #pragma unroll
    for (int r = 0; r < 64; ++r) s += evr[r] * W[r * 64 + tid];
    ew[tid] = s;
  } else if (tid == 64) {
    float s = 0.f;
    for (int r = 0; r < 64; ++r) s += evr[r] * bml[r];
    Cc[tau] = s;
  }
  __syncthreads();
  float part = 0.f;
  for (int t = tid; t < KPAD; t += 256) {
    float s = 0.f;
    if (t < SEQ) {
      const float* v = V + (size_t)(SEQ - 1 - t) * 64;
      #pragma unroll
      for (int k = 0; k < 64; ++k) s += ew[k] * v[k];
      part += s;
    }
    KK[(size_t)tau * KPAD + t] = (_Float16)s;
  }
  red[tid] = part;
  __syncthreads();
  for (int off = 128; off > 0; off >>= 1) {
    if (tid < off) red[tid] += red[tid + off];
    __syncthreads();
  }
  if (tid == 0) R[tau] = red[0];
}

// ------- kernel 5: main WMMA GEMM + fused (de)normalization epilogue -------
// one wave per block; wave tile = 64 rows (p) x 48 cols (tau); K = 736 in chunks of 32
__global__ __launch_bounds__(32) void gemm_kernel(
    const _Float16* __restrict__ X, const _Float16* __restrict__ KK,
    const float* __restrict__ stats, const float* __restrict__ R,
    const float* __restrict__ Cc, float* __restrict__ out) {
  __shared__ float lds[48][65];
  const int lane = threadIdx.x;
  const int half = lane >> 4, ln = lane & 15;
  const int pbase = blockIdx.x * 64;
  const int tbase = blockIdx.y * 48;

  v8f acc[4][3] = {};

  const _Float16* arow[4];
  const _Float16* brow[3];
  #pragma unroll
  for (int mt = 0; mt < 4; ++mt)
    arow[mt] = X + (size_t)(pbase + 16 * mt + ln) * KPAD + 8 * half;   // A-op: lane = row p
  #pragma unroll
  for (int nt = 0; nt < 3; ++nt)
    brow[nt] = KK + (size_t)(tbase + 16 * nt + ln) * KPAD + 8 * half;  // B-op: lane = col tau

  for (int kt = 0; kt < KPAD; kt += 32) {
    v16h a[4], b[3];
    #pragma unroll
    for (int mt = 0; mt < 4; ++mt) {
      v8h lo = *reinterpret_cast<const v8h*>(arow[mt] + kt);       // K = kt + 8*half + 0..7
      v8h hi = *reinterpret_cast<const v8h*>(arow[mt] + kt + 16);  // K = kt + 16 + 8*half + 0..7
      a[mt] = __builtin_shufflevector(lo, hi, 0,1,2,3,4,5,6,7,8,9,10,11,12,13,14,15);
    }
    #pragma unroll
    for (int nt = 0; nt < 3; ++nt) {
      v8h lo = *reinterpret_cast<const v8h*>(brow[nt] + kt);       // K = kt + 8*half + 0..7
      v8h hi = *reinterpret_cast<const v8h*>(brow[nt] + kt + 16);  // K = kt + 16 + 8*half + 0..7
      b[nt] = __builtin_shufflevector(lo, hi, 0,1,2,3,4,5,6,7,8,9,10,11,12,13,14,15);
    }
    #pragma unroll
    for (int mt = 0; mt < 4; ++mt) {
      #pragma unroll
      for (int nt = 0; nt < 3; ++nt) {
        acc[mt][nt] = __builtin_amdgcn_wmma_f32_16x16x32_f16(
            false, a[mt], false, b[nt], (short)0, acc[mt][nt], false, false);
      }
    }
  }

  // epilogue: pre = alpha*acc + beta*R[tau] + c[tau]; out = (pre-ab)*iaw*std + mean
  float rv[3], cv[3];
  #pragma unroll
  for (int nt = 0; nt < 3; ++nt) {
    int tau = tbase + 16 * nt + ln;
    rv[nt] = R[tau]; cv[nt] = Cc[tau];
  }
  #pragma unroll
  for (int mt = 0; mt < 4; ++mt) {
    #pragma unroll
    for (int r = 0; r < 8; ++r) {
      int p = pbase + 16 * mt + 8 * half + r;
      int ps = (p < NP) ? p : (NP - 1);
      const float* s = stats + (size_t)ps * 8;
      float mean = s[0], sd = s[1], alpha = s[2], beta = s[3], abv = s[4], iaw = s[5];
      #pragma unroll
      for (int nt = 0; nt < 3; ++nt) {
        float val = alpha * acc[mt][nt][r] + beta * rv[nt] + cv[nt];
        float o = (val - abv) * iaw * sd + mean;
        lds[16 * nt + ln][16 * mt + 8 * half + r] = o;
      }
    }
  }
  __syncthreads();
  for (int tl = 0; tl < 48; ++tl) {
    int tau = tbase + tl;
    #pragma unroll
    for (int e = 0; e < 2; ++e) {
      int pl = lane + 32 * e;
      int p = pbase + pl;
      if (p < NP) {
        int bb = p / CH;
        int c = p - bb * CH;
        out[((size_t)bb * SEQ + tau) * CH + c] = lds[tl][pl];
      }
    }
  }
}

extern "C" void kernel_launch(void* const* d_in, const int* in_sizes, int n_in,
                              void* d_out, int out_size, void* d_ws, size_t ws_size,
                              hipStream_t stream) {
  const float* x   = (const float*)d_in[0];  // (16,720,321)
  const float* A   = (const float*)d_in[1];  // (64,64)
  const float* Bv  = (const float*)d_in[2];  // (64,)
  const float* ev  = (const float*)d_in[3];  // (720,64)
  const float* W   = (const float*)d_in[4];  // (64,64)
  const float* bml = (const float*)d_in[5];  // (64,)
  const float* aw  = (const float*)d_in[6];  // (1,1,321)
  const float* ab  = (const float*)d_in[7];  // (1,1,321)
  float* out = (float*)d_out;

  char* ws = (char*)d_ws;
  float*    stats = (float*)(ws + OFF_STATS);
  float*    V     = (float*)(ws + OFF_V);
  float*    R     = (float*)(ws + OFF_R);
  float*    Cc    = (float*)(ws + OFF_C);
  _Float16* KK    = (_Float16*)(ws + OFF_KK);
  _Float16* X16   = (_Float16*)(ws + OFF_X);

  stats_kernel <<<dim3(16, 2), 256, 0, stream>>>(x, aw, ab, stats);
  xcast_kernel <<<dim3(16, 23, 6), 256, 0, stream>>>(x, X16);
  krylov_kernel<<<1, 1024, 0, stream>>>(A, Bv, V);
  kk_kernel    <<<720, 256, 0, stream>>>(ev, W, bml, V, KK, R, Cc);
  gemm_kernel  <<<dim3(81, 15), 32, 0, stream>>>(X16, KK, stats, R, Cc, out);
}
